// QuerySAT_40123584479548
// MI455X (gfx1250) — matrix operand order repaired and checked by reading, not compiled, not run on hardware
//
#include <hip/hip_runtime.h>
#include <hip/hip_bf16.h>

typedef __attribute__((ext_vector_type(16))) __bf16 v16bf;
typedef __attribute__((ext_vector_type(8)))  float  v8f;

// gcc-style int4 vector matching the async-LDS builtin's pointee type
typedef int v4i_vs __attribute__((vector_size(16)));
typedef __attribute__((address_space(1))) v4i_vs as1_v4i;  // global
typedef __attribute__((address_space(3))) v4i_vs as3_v4i;  // LDS

#define N_VARS    8192
#define N_CLAUSES 32768
#define N_LITS    (3 * N_CLAUSES)
#define ROUNDS    16

// LDS row strides in bf16 elements (row byte stride 80 = 16B-aligned, padded)
#define AS 40
#define BS 40

#if __has_builtin(__builtin_amdgcn_global_load_async_to_lds_b128)
#define HAVE_ASYNC_LDS 1
#endif

__device__ __forceinline__ float leaky_f(float x)   { return x > 0.f ? x : 0.2f * x; }
__device__ __forceinline__ float softplus_f(float x){ return (x > 20.f) ? x : log1pf(expf(x)); }
__device__ __forceinline__ float sigmoid_f(float x) { return 1.f / (1.f + expf(-x)); }

__device__ __forceinline__ float waveSum(float v) {
#pragma unroll
    for (int o = 16; o > 0; o >>= 1) v += __shfl_xor(v, o, 32);
    return v;
}

// 16-byte global -> LDS copy; async (ASYNCcnt) when available.
__device__ __forceinline__ void cp16_g2l(const __bf16* g, __bf16* l) {
#ifdef HAVE_ASYNC_LDS
    void* gv = (void*)g;   // drop const
    void* lv = (void*)l;
    __builtin_amdgcn_global_load_async_to_lds_b128(
        (as1_v4i*)gv, (as3_v4i*)lv, 0, 0);
#else
    *reinterpret_cast<uint4*>(l) = *reinterpret_cast<const uint4*>(g);
#endif
}

__device__ __forceinline__ void wait_async_lds() {
#ifdef HAVE_ASYNC_LDS
#if __has_builtin(__builtin_amdgcn_s_wait_asynccnt)
    __builtin_amdgcn_s_wait_asynccnt(0);
#else
    asm volatile("s_wait_asynccnt 0x0" ::: "memory");
#endif
#endif
}

// ---------------------------------------------------------------------------
// Tiled GEMM: C[M,N] = act(A[M,K] * W[K,N] + bias), W pre-transposed bf16 [N,K].
// Block 256 threads = 8 waves; block tile 128x64; wave tile 32x32 as 2x2
// v_wmma_f32_16x16x32_bf16. Double-buffered LDS; bf16 tiles staged via
// GLOBAL_LOAD_ASYNC_TO_LDS_B128 (overlapped with WMMA), f32 A converted on stage.
// ABF16: A is bf16 (else f32). OBF16: store bf16 output (else f32).
// ---------------------------------------------------------------------------
template <int ACT, int ABF16, int OBF16>
__global__ __launch_bounds__(256) void gemm_k(
    const void* __restrict__ Aptr, int lda,
    const __bf16* __restrict__ Wt,          // [N,K] row-major
    const float* __restrict__ bias,
    void* __restrict__ Cptr, int ldc, int K)
{
    __shared__ __bf16 Abuf[2][128 * AS];
    __shared__ __bf16 Bbuf[2][64 * BS];

    const int tid     = threadIdx.x;
    const int m_block = blockIdx.y * 128;
    const int n_block = blockIdx.x * 64;
    const int wave    = tid >> 5;
    const int lane    = tid & 31;
    const int half    = lane >> 4;
    const int l16     = lane & 15;
    const int m_wave  = (wave & 3) * 32;
    const int n_wave  = (wave >> 2) * 32;

    const int arow = tid >> 1;          // 128 rows, 2 threads/row
    const int acol = (tid & 1) * 16;    // 16 elements each
    const int brow = tid >> 2;          // 64 rows, 4 threads/row
    const int bcol = (tid & 3) * 8;     // 8 bf16 each

    v8f acc[2][2] = {};

    auto stageA = [&](int buf, int k0) {
        if constexpr (ABF16) {
            const __bf16* A = (const __bf16*)Aptr;
            const __bf16* g = A + (size_t)(m_block + arow) * lda + k0 + acol;
            __bf16* l = &Abuf[buf][arow * AS + acol];
            cp16_g2l(g, l);
            cp16_g2l(g + 8, l + 8);
        } else {
            const float* A = (const float*)Aptr;
            const float4* ap4 =
                reinterpret_cast<const float4*>(A + (size_t)(m_block + arow) * lda + k0 + acol);
            float4 q0 = ap4[0], q1 = ap4[1], q2 = ap4[2], q3 = ap4[3];
            float vv[16] = {q0.x, q0.y, q0.z, q0.w, q1.x, q1.y, q1.z, q1.w,
                            q2.x, q2.y, q2.z, q2.w, q3.x, q3.y, q3.z, q3.w};
            __bf16* d = &Abuf[buf][arow * AS + acol];
#pragma unroll
            for (int i = 0; i < 16; ++i) d[i] = (__bf16)vv[i];
        }
    };
    auto stageB = [&](int buf, int k0) {
        cp16_g2l(Wt + (size_t)(n_block + brow) * K + k0 + bcol,
                 &Bbuf[buf][brow * BS + bcol]);
    };

    stageA(0, 0);
    stageB(0, 0);
    wait_async_lds();
    __syncthreads();

    for (int k0 = 0; k0 < K; k0 += 32) {
        const int buf = (k0 >> 5) & 1;
        if (k0 + 32 < K) {              // prefetch next tile while computing
            stageA(buf ^ 1, k0 + 32);
            stageB(buf ^ 1, k0 + 32);
        }

        // fragments per ISA layout
        v16bf afrag[2], bfrag[2];
#pragma unroll
        for (int mi = 0; mi < 2; ++mi) {
            const __bf16* p = &Abuf[buf][(m_wave + mi * 16 + l16) * AS];
#pragma unroll
            for (int i = 0; i < 8; ++i) afrag[mi][i] = p[half * 8 + i];          // K 0-7 / 8-15
#pragma unroll
            for (int i = 0; i < 8; ++i) afrag[mi][8 + i] = p[16 + half * 8 + i]; // K 16-23 / 24-31
        }
#pragma unroll
        for (int ni = 0; ni < 2; ++ni) {
            const __bf16* p = &Bbuf[buf][(n_wave + ni * 16 + l16) * BS];
#pragma unroll
            for (int i = 0; i < 8; ++i) { bfrag[ni][i] = p[half * 8 + i]; bfrag[ni][8 + i] = p[16 + half * 8 + i]; }
        }
#pragma unroll
        for (int mi = 0; mi < 2; ++mi)
#pragma unroll
            for (int ni = 0; ni < 2; ++ni)
                acc[mi][ni] = __builtin_amdgcn_wmma_f32_16x16x32_bf16(
                    false, afrag[mi], false, bfrag[ni], (short)0, acc[mi][ni], false, false);

        wait_async_lds();
        __syncthreads();
    }

    // epilogue (C/D layout: VGPR r -> M = r + 8*half)
#pragma unroll
    for (int ni = 0; ni < 2; ++ni) {
        const int gn = n_block + n_wave + ni * 16 + l16;
        const float bv = bias[gn];
#pragma unroll
        for (int mi = 0; mi < 2; ++mi) {
#pragma unroll
            for (int r = 0; r < 8; ++r) {
                const int gm = m_block + m_wave + mi * 16 + half * 8 + r;
                float v = acc[mi][ni][r] + bv;
                if (ACT) v = leaky_f(v);
                if constexpr (OBF16)
                    ((__bf16*)Cptr)[(size_t)gm * ldc + gn] = (__bf16)v;
                else
                    ((float*)Cptr)[(size_t)gm * ldc + gn] = v;
            }
        }
    }
}

// Weight transpose + f32->bf16: Wt[n*K+k] = W[k*N+n]
__global__ void wconv_k(const float* __restrict__ W, __bf16* __restrict__ Wt, int K, int N)
{
    int idx = blockIdx.x * 256 + threadIdx.x;
    if (idx < K * N) {
        int n = idx / K;
        int k = idx - n * K;
        Wt[idx] = (__bf16)W[(size_t)k * N + n];
    }
}

// zero_state init: col 0 = 3.984375, else -0.015625  ((onehot - 1/256) * 16 * 0.25)
__global__ void init_state_k(float* __restrict__ dst, int ld, int rows)
{
    int idx = blockIdx.x * 256 + threadIdx.x;
    if (idx < rows * 256) {
        int r = idx >> 8, f = idx & 255;
        dst[(size_t)r * ld + f] = (f == 0) ? 3.984375f : -0.015625f;
    }
}

// zero unit columns 256..447 (grad/vlp/vln accumulators)
__global__ void zero_grad_k(float* __restrict__ unit)
{
    int idx = blockIdx.x * 256 + threadIdx.x;
    if (idx < N_VARS * 192) {
        int r = idx / 192, c = idx - r * 192;
        unit[(size_t)r * 448 + 256 + c] = 0.f;
    }
}

// Clause values + analytic gradient (lit_clause = repeat(arange,3): contiguous).
__global__ void clause_k(const float* __restrict__ query,
                         const int* __restrict__ lit_vars,
                         const int* __restrict__ lit_sign_bits,
                         float* __restrict__ clause_unit,
                         float* __restrict__ unit)
{
    int idx = blockIdx.x * 256 + threadIdx.x;
    if (idx >= N_CLAUSES * 64) return;
    int c = idx >> 6, f = idx & 63;

    float x[3], sg[3];
    int vs[3];
    float s = 0.f;
#pragma unroll
    for (int j = 0; j < 3; ++j) {
        int l = 3 * c + j;
        int v = lit_vars[l];
        float sign = lit_sign_bits[l] ? 1.f : -1.f;
        float xv = sign * query[(size_t)v * 64 + f];
        vs[j] = v; sg[j] = sign; x[j] = xv;
        s += softplus_f(xv);
    }
    float cl = expf(-s);
    clause_unit[(size_t)c * 320 + 256 + f] = cl;
#pragma unroll
    for (int j = 0; j < 3; ++j) {
        float g = -cl * sigmoid_f(x[j]) * sg[j];
        atomicAdd(&unit[(size_t)vs[j] * 448 + 256 + f], g);
    }
}

// vlp/vln literal scatter: pos -> unit[:,320+f], neg -> unit[:,384+f]
__global__ void scatter_k(const float* __restrict__ clause_data, // [N_CLAUSES,384]
                          const int* __restrict__ lit_vars,
                          const int* __restrict__ lit_sign_bits,
                          float* __restrict__ unit)
{
    int idx = blockIdx.x * 256 + threadIdx.x;
    if (idx >= N_LITS * 64) return;
    int l = idx >> 6, f = idx & 63;
    int c = l / 3;
    int v = lit_vars[l];
    if (lit_sign_bits[l]) {
        atomicAdd(&unit[(size_t)v * 448 + 320 + f], clause_data[(size_t)c * 384 + f]);
    } else {
        atomicAdd(&unit[(size_t)v * 448 + 384 + f], clause_data[(size_t)c * 384 + 64 + f]);
    }
}

// LayerNorm over 256 cols, *0.25; one wave per row.
__global__ __launch_bounds__(256) void layernorm256_k(
    const float* __restrict__ in, int in_ld,
    float* __restrict__ out, int out_ld,
    const float* __restrict__ g, const float* __restrict__ b, int rows)
{
    int wave = threadIdx.x >> 5, lane = threadIdx.x & 31;
    int row = blockIdx.x * 8 + wave;
    if (row >= rows) return;
    const float* ip = in + (size_t)row * in_ld;
    float x[8], s = 0.f, s2 = 0.f;
#pragma unroll
    for (int i = 0; i < 8; ++i) {
        float v = ip[i * 32 + lane];
        x[i] = v; s += v; s2 += v * v;
    }
    s = waveSum(s);
    s2 = waveSum(s2);
    float mu = s * (1.f / 256.f);
    float var = s2 * (1.f / 256.f) - mu * mu;
    float r = rsqrtf(var + 1e-3f);
    float* op = out + (size_t)row * out_ld;
#pragma unroll
    for (int i = 0; i < 8; ++i) {
        int cidx = i * 32 + lane;
        op[cidx] = ((x[i] - mu) * r * g[cidx] + b[cidx]) * 0.25f;
    }
}

// Final o-MLP layer (N=1 GEMV): one wave per row.
__global__ __launch_bounds__(256) void gemv256_k(
    const float* __restrict__ A, const float* __restrict__ w,
    const float* __restrict__ b, float* __restrict__ out, int rows)
{
    int wave = threadIdx.x >> 5, lane = threadIdx.x & 31;
    int row = blockIdx.x * 8 + wave;
    if (row >= rows) return;
    const float* ap = A + (size_t)row * 256;
    float s = 0.f;
#pragma unroll
    for (int i = 0; i < 8; ++i) s += ap[i * 32 + lane] * w[i * 32 + lane];
    s = waveSum(s);
    if (lane == 0) out[row] = s + b[0];
}

// ---------------------------------------------------------------------------
extern "C" void kernel_launch(void* const* d_in, const int* in_sizes, int n_in,
                              void* d_out, int out_size, void* d_ws, size_t ws_size,
                              hipStream_t stream)
{
    (void)in_sizes; (void)n_in; (void)out_size; (void)ws_size;

    const float* qw[3] = {(const float*)d_in[0],  (const float*)d_in[2],  (const float*)d_in[4]};
    const float* qb[3] = {(const float*)d_in[1],  (const float*)d_in[3],  (const float*)d_in[5]};
    const float* cw[3] = {(const float*)d_in[6],  (const float*)d_in[8],  (const float*)d_in[10]};
    const float* cb[3] = {(const float*)d_in[7],  (const float*)d_in[9],  (const float*)d_in[11]};
    const float* uw[3] = {(const float*)d_in[12], (const float*)d_in[14], (const float*)d_in[16]};
    const float* ub[3] = {(const float*)d_in[13], (const float*)d_in[15], (const float*)d_in[17]};
    const float* ow[3] = {(const float*)d_in[18], (const float*)d_in[20], (const float*)d_in[22]};
    const float* ob[3] = {(const float*)d_in[19], (const float*)d_in[21], (const float*)d_in[23]};
    const float* vn_g = (const float*)d_in[24];
    const float* vn_b = (const float*)d_in[25];
    const float* cn_g = (const float*)d_in[26];
    const float* cn_b = (const float*)d_in[27];
    const int* lit_vars  = (const int*)d_in[28];
    const int* lit_signs = (const int*)d_in[29];
    float* out = (float*)d_out;

    // Arena: unit[8192,448] f32 | clause_unit[32768,320] f32 | fbuf[32768,384] f32
    //        | bbufA, bbufB (bf16, 32768*768 each) | bf16 weight arena
    float* ws = (float*)d_ws;
    const size_t UNIT = 0;
    const size_t CLU  = UNIT + (size_t)N_VARS * 448;
    const size_t FBUF = CLU + (size_t)N_CLAUSES * 320;
    const size_t BB_A = FBUF + (size_t)N_CLAUSES * 384;            // in floats
    const size_t BB_B = BB_A + (size_t)N_CLAUSES * 768 / 2;        // bf16 = half a float
    const size_t WOFF = BB_B + (size_t)N_CLAUSES * 768 / 2;
    float*  unit        = ws + UNIT;
    float*  clause_unit = ws + CLU;
    float*  fbuf        = ws + FBUF;
    __bf16* bbufA       = (__bf16*)(ws + BB_A);
    __bf16* bbufB       = (__bf16*)(ws + BB_B);
    __bf16* warena      = (__bf16*)(ws + WOFF);

    struct WSpec { const float* w; int K, N; };
    WSpec specs[11] = {
        {qw[0], 256, 128}, {qw[1], 128, 128}, {qw[2], 128, 64},
        {cw[0], 320, 768}, {cw[1], 768, 768}, {cw[2], 768, 384},
        {uw[0], 448, 512}, {uw[1], 512, 512}, {uw[2], 512, 256},
        {ow[0], 256, 256}, {ow[1], 256, 256}};
    __bf16* wt[11];
    {
        size_t off = 0;
        for (int i = 0; i < 11; ++i) {
            wt[i] = warena + off;
            off += (size_t)specs[i].K * specs[i].N;
        }
    }
    for (int i = 0; i < 11; ++i) {
        int total = specs[i].K * specs[i].N;
        wconv_k<<<(total + 255) / 256, 256, 0, stream>>>(specs[i].w, wt[i],
                                                         specs[i].K, specs[i].N);
    }

    init_state_k<<<(N_VARS * 256 + 255) / 256, 256, 0, stream>>>(unit, 448, N_VARS);
    init_state_k<<<(N_CLAUSES * 256 + 255) / 256, 256, 0, stream>>>(clause_unit, 320, N_CLAUSES);

    // act: leaky? | abf: A is bf16? | obf: output bf16?
    auto gemm = [&](const void* A, int lda, const __bf16* Wt, const float* bias,
                    void* C, int ldc, int M, int N, int K, int act, int abf, int obf) {
        dim3 g(N / 64, M / 128);
        if (act && !abf && obf)      gemm_k<1, 0, 1><<<g, 256, 0, stream>>>(A, lda, Wt, bias, C, ldc, K);
        else if (act && abf && obf)  gemm_k<1, 1, 1><<<g, 256, 0, stream>>>(A, lda, Wt, bias, C, ldc, K);
        else if (act && abf && !obf) gemm_k<1, 1, 0><<<g, 256, 0, stream>>>(A, lda, Wt, bias, C, ldc, K);
        else                         gemm_k<0, 1, 0><<<g, 256, 0, stream>>>(A, lda, Wt, bias, C, ldc, K);
    };

    for (int r = 0; r < ROUNDS; ++r) {
        // q-MLP: unit[:, :256] (f32) -> bf16 -> bf16 -> query f32 [8192,64]
        gemm(unit, 448, wt[0], qb[0], bbufA, 128, N_VARS, 128, 256, 1, 0, 1);
        gemm(bbufA, 128, wt[1], qb[1], bbufB, 128, N_VARS, 128, 128, 1, 1, 1);
        gemm(bbufB, 128, wt[2], qb[2], fbuf, 64, N_VARS, 64, 128, 0, 1, 0);

        zero_grad_k<<<(N_VARS * 192 + 255) / 256, 256, 0, stream>>>(unit);
        clause_k<<<(N_CLAUSES * 64 + 255) / 256, 256, 0, stream>>>(
            fbuf, lit_vars, lit_signs, clause_unit, unit);

        // c-MLP: clause_unit f32 [32768,320] -> bf16 -> bf16 -> clause_data f32 [32768,384]
        gemm(clause_unit, 320, wt[3], cb[0], bbufA, 768, N_CLAUSES, 768, 320, 1, 0, 1);
        gemm(bbufA, 768, wt[4], cb[1], bbufB, 768, N_CLAUSES, 768, 768, 1, 1, 1);
        gemm(bbufB, 768, wt[5], cb[2], fbuf, 384, N_CLAUSES, 384, 768, 0, 1, 0);

        scatter_k<<<(N_LITS * 64 + 255) / 256, 256, 0, stream>>>(
            fbuf, lit_vars, lit_signs, unit);
        layernorm256_k<<<N_CLAUSES / 8, 256, 0, stream>>>(
            fbuf + 128, 384, clause_unit, 320, cn_g, cn_b, N_CLAUSES);

        // u-MLP: unit f32 [8192,448] -> bf16 -> bf16 -> f32 [8192,256]; LN -> unit[:, :256]
        gemm(unit, 448, wt[6], ub[0], bbufA, 512, N_VARS, 512, 448, 1, 0, 1);
        gemm(bbufA, 512, wt[7], ub[1], bbufB, 512, N_VARS, 512, 512, 1, 1, 1);
        gemm(bbufB, 512, wt[8], ub[2], fbuf, 256, N_VARS, 256, 512, 0, 1, 0);
        layernorm256_k<<<N_VARS / 8, 256, 0, stream>>>(
            fbuf, 256, unit, 448, vn_g, vn_b, N_VARS);

        // o-MLP -> logits
        gemm(unit, 448, wt[9], ob[0], bbufA, 256, N_VARS, 256, 256, 1, 0, 1);
        gemm(bbufA, 256, wt[10], ob[1], fbuf, 256, N_VARS, 256, 256, 1, 1, 0);
        gemv256_k<<<N_VARS / 8, 256, 0, stream>>>(
            fbuf, ow[2], ob[2], out + (size_t)r * N_VARS, N_VARS);
    }
}
